// W8A8BFP32OFP32Linear_32968168964166
// MI455X (gfx1250) — 8-bit matrix operand order repaired and checked by reading, compiled, not run on hardware
//
#include <hip/hip_runtime.h>
#include <stdint.h>

typedef __attribute__((ext_vector_type(8))) int v8i;

#define K_DIM 4096
#define N_DIM 4096
#define TILE_K 64
#define WG_M 128
#define WG_N 128
#define K_STEPS (K_DIM / TILE_K)
#define ABUF (WG_M * (TILE_K / 4))
#define BBUF (WG_N * (TILE_K / 4))

__global__ __launch_bounds__(256) void w8a8_i8gemm_wmma(
    const int8_t* __restrict__ X,
    const int8_t* __restrict__ W,
    const float*  __restrict__ bias,
    const float*  __restrict__ alphap,
    float*        __restrict__ Y)
{
    __shared__ __align__(16) int ldsA[2 * ABUF];
    __shared__ __align__(16) int ldsB[2 * BBUF];

    const int tid  = threadIdx.x;
    const int lane = tid & 31;
    const int wid  = tid >> 5;
    const int waveM = wid & 3;
    const int waveN = wid >> 2;

    const int m0 = blockIdx.y * WG_M;
    const int n0 = blockIdx.x * WG_N;

    const float alpha = alphap[0];

    const int row0 = tid >> 2;
    const int row1 = row0 + 64;
    const int chk  = tid & 3;
    const int4* gA0 = (const int4*)(X + (size_t)(m0 + row0) * K_DIM) + chk;
    const int4* gA1 = (const int4*)(X + (size_t)(m0 + row1) * K_DIM) + chk;
    const int4* gB0 = (const int4*)(W + (size_t)(n0 + row0) * K_DIM) + chk;
    const int4* gB1 = (const int4*)(W + (size_t)(n0 + row1) * K_DIM) + chk;

    int4 rA0 = gA0[0], rA1 = gA1[0], rB0 = gB0[0], rB1 = gB1[0];
    ((int4*)ldsA)[row0 * 4 + chk] = rA0;
    ((int4*)ldsA)[row1 * 4 + chk] = rA1;
    ((int4*)ldsB)[row0 * 4 + chk] = rB0;
    ((int4*)ldsB)[row1 * 4 + chk] = rB1;
    __syncthreads();

    const v8i vzero = {};
    v8i acc[2][4];
#pragma unroll
    for (int mi = 0; mi < 2; ++mi)
#pragma unroll
        for (int ni = 0; ni < 4; ++ni) acc[mi][ni] = vzero;

    const int hi  = lane >> 4;
    const int r16 = lane & 15;
    const int mBase = waveM * 32;
    const int nBase = waveN * 64;

#pragma unroll 1
    for (int kt = 0; kt < K_STEPS; ++kt) {
        const int  cur  = kt & 1;
        const int* bufA = ldsA + cur * ABUF;
        const int* bufB = ldsB + cur * BBUF;
        const bool more = (kt + 1 < K_STEPS);

        if (more) {
            gA0 += 4; gA1 += 4; gB0 += 4; gB1 += 4;
            rA0 = gA0[0]; rA1 = gA1[0]; rB0 = gB0[0]; rB1 = gB1[0];
        }

        v8i aF[2], bF[4];
#pragma unroll
        for (int v = 0; v < 8; ++v) {
            const int kA = ((v >> 1) << 2) + (v & 1) + (hi << 1);
            aF[0][v] = bufA[(mBase +      r16) * 16 + kA];
            aF[1][v] = bufA[(mBase + 16 + r16) * 16 + kA];
            const int kB = ((v >> 1) << 2) + (v & 1) + (hi << 1);
            bF[0][v] = bufB[(nBase +      r16) * 16 + kB];
            bF[1][v] = bufB[(nBase + 16 + r16) * 16 + kB];
            bF[2][v] = bufB[(nBase + 32 + r16) * 16 + kB];
            bF[3][v] = bufB[(nBase + 48 + r16) * 16 + kB];
        }

#pragma unroll
        for (int mi = 0; mi < 2; ++mi)
#pragma unroll
            for (int ni = 0; ni < 4; ++ni)
                acc[mi][ni] = __builtin_amdgcn_wmma_i32_16x16x64_iu8(
                    true, aF[mi], true, bF[ni], acc[mi][ni], false, false);

        if (more) {
            const int nxt = cur ^ 1;
            ((int4*)(ldsA + nxt * ABUF))[row0 * 4 + chk] = rA0;
            ((int4*)(ldsA + nxt * ABUF))[row1 * 4 + chk] = rA1;
            ((int4*)(ldsB + nxt * BBUF))[row0 * 4 + chk] = rB0;
            ((int4*)(ldsB + nxt * BBUF))[row1 * 4 + chk] = rB1;
        }
        __syncthreads();
    }

    const int gm = m0 + mBase;
    const int gn = n0 + nBase;
    float bv[4];
#pragma unroll
    for (int ni = 0; ni < 4; ++ni) bv[ni] = bias[gn + ni * 16 + r16];

#pragma unroll
    for (int r = 0; r < 8; ++r) {
        const int Mr = r + (hi << 3);
#pragma unroll
        for (int mi = 0; mi < 2; ++mi) {
            float* yrow = Y + (size_t)(gm + mi * 16 + Mr) * N_DIM;
#pragma unroll
            for (int ni = 0; ni < 4; ++ni)
                yrow[gn + ni * 16 + r16] = alpha * (float)acc[mi][ni][r] + bv[ni];
        }
    }
}

extern "C" void kernel_launch(void* const* d_in, const int* in_sizes, int n_in,
                              void* d_out, int out_size, void* d_ws, size_t ws_size,
                              hipStream_t stream) {
    const int8_t* x     = (const int8_t*)d_in[0];
    const int8_t* w     = (const int8_t*)d_in[1];
    const float*  bias  = (const float*)d_in[2];
    const float*  alpha = (const float*)d_in[3];
    float* y = (float*)d_out;

    const int M = in_sizes[0] / K_DIM;
    dim3 grid(N_DIM / WG_N, M / WG_M);
    dim3 block(256);
    w8a8_i8gemm_wmma<<<grid, block, 0, stream>>>(x, w, bias, alpha, y);
}
